// PairPaths2Embed_9689446220166
// MI455X (gfx1250) — compile-verified, matrix operand order checked by reading, not run on hardware
//
#include <hip/hip_runtime.h>
#include <hip/hip_bf16.h>

// ---------------------------------------------------------------------------
// MI455X (gfx1250) fused implementation, round 2.
//  - All matmuls via v_wmma_f32_16x16x32_f16 (f32 accumulate).
//  - One block (256 thr = 8 wave32) owns a 16-node tile; full per-node chain
//    fused, intermediates in LDS.
//  - Global pointers annotated via is_shared/is_private assumes so weight /
//    bias / feature traffic uses global_load_* (not flat_*).
//  - Feature tiles staged to LDS with global_load_async_to_lds_b128
//    (ASYNCcnt) then converted f32->f16 in LDS.
// ---------------------------------------------------------------------------

typedef __attribute__((ext_vector_type(16))) _Float16 v16h;
typedef __attribute__((ext_vector_type(8)))  _Float16 v8h;
typedef __attribute__((ext_vector_type(8)))  float    v8f;

#define DEV __device__ __forceinline__

DEV int lane_id() { return threadIdx.x & 31; }

template <typename T>
DEV T* as_global(T* p) {
  __builtin_assume(!__builtin_amdgcn_is_shared((const void*)p) &&
                   !__builtin_amdgcn_is_private((const void*)p));
  return p;
}

// A tile (16x32 f16) from LDS. CDNA5 ISA 7.12.2 layout: lanes 0-15 rows M,
// halves 0-7 = K k0+kh.., halves 8-15 = K k0+16+kh.., kh = 8*(lane>=16).
DEV v16h load_a16(const _Float16* A, int lda, int k0) {
  const int l = lane_id();
  const _Float16* p = A + (l & 15) * lda + k0 + ((l >> 4) << 3);
  v8h lo = *(const v8h*)p;
  v8h hi = *(const v8h*)(p + 16);
  return __builtin_shufflevector(lo, hi, 0,1,2,3,4,5,6,7,8,9,10,11,12,13,14,15);
}

// B tile (32x16 f16) from f16 weight stored as W[fout][fin] (B = W^T with
// contiguous K). Mirrors A layout with N in place of M.
DEV v16h load_b16(const _Float16* W, int K, int n0, int k0) {
  const int l = lane_id();
  const _Float16* p = W + (size_t)(n0 + (l & 15)) * K + k0 + ((l >> 4) << 3);
  v8h lo = *(const v8h*)p;
  v8h hi = *(const v8h*)(p + 16);
  return __builtin_shufflevector(lo, hi, 0,1,2,3,4,5,6,7,8,9,10,11,12,13,14,15);
}

template <int K>
DEV v8f gemm16(const _Float16* A, int lda, const _Float16* W, int n0) {
  v8f acc = {0.f, 0.f, 0.f, 0.f, 0.f, 0.f, 0.f, 0.f};
#pragma unroll 4
  for (int k0 = 0; k0 < K; k0 += 32) {
    v16h a = load_a16(A, lda, k0);
    v16h b = load_b16(W, K, n0, k0);
    acc = __builtin_amdgcn_wmma_f32_16x16x32_f16(false, a, false, b,
                                                 (short)0, acc, false, false);
  }
  return acc;
}

// C/D layout: lane n = n0 + (l&15); vgpr r -> row m = r + 8*(l>=16).
DEV void store_f32(float* C, int ldc, int n0, v8f acc, const float* bias) {
  const int l = lane_id(); const int n = n0 + (l & 15); const int mb = (l >> 4) << 3;
  const float bv = bias[n];
#pragma unroll
  for (int r = 0; r < 8; ++r) C[(mb + r) * ldc + n] = acc[r] + bv;
}

DEV void store_f32_res(float* C, int ldc, int n0, v8f acc, const float* bias,
                       const _Float16* R, int ldr) {
  const int l = lane_id(); const int n = n0 + (l & 15); const int mb = (l >> 4) << 3;
  const float bv = bias[n];
#pragma unroll
  for (int r = 0; r < 8; ++r)
    C[(mb + r) * ldc + n] = acc[r] + bv + (float)R[(mb + r) * ldr + n];
}

DEV void store_f16(_Float16* O, int ldo, int n0, v8f acc, const float* bias, bool relu) {
  const int l = lane_id(); const int n = n0 + (l & 15); const int mb = (l >> 4) << 3;
  const float bv = bias[n];
#pragma unroll
  for (int r = 0; r < 8; ++r) {
    float v = acc[r] + bv;
    if (relu) v = fmaxf(v, 0.f);
    O[(mb + r) * ldo + n] = (_Float16)v;
  }
}

// ---- LayerNorm over a 16x128 f32 tile (256 threads), branch-free variants --

DEV float ln_stats(const float* C, int ldc, float* red, int row, int seg,
                   float& mu_out) {
  const float* Cr = C + row * ldc + seg * 8;
  float s = 0.f, ss = 0.f;
#pragma unroll
  for (int j = 0; j < 8; ++j) { float v = Cr[j]; s += v; ss += v * v; }
  red[row * 16 + seg] = s; red[256 + row * 16 + seg] = ss;
  __syncthreads();
  float mu = 0.f, m2 = 0.f;
#pragma unroll
  for (int j = 0; j < 16; ++j) { mu += red[row * 16 + j]; m2 += red[256 + row * 16 + j]; }
  mu *= (1.f / 128.f); m2 = m2 * (1.f / 128.f) - mu * mu;
  mu_out = mu;
  return rsqrtf(m2 + 1e-5f);
}

DEV void ln_tile_lds(const float* C, int ldc, const float* w, const float* b,
                     _Float16* o1, int ld1, bool do_tanh, float* red) {
  const int row = threadIdx.x >> 4, seg = threadIdx.x & 15;
  float mu; const float inv = ln_stats(C, ldc, red, row, seg, mu);
#pragma unroll
  for (int j = 0; j < 8; ++j) {
    const int c = seg * 8 + j;
    float v = (C[row * ldc + c] - mu) * inv * w[c] + b[c];
    if (do_tanh) v = tanhf(v);
    o1[row * ld1 + c] = (_Float16)v;
  }
  __syncthreads();
}

DEV void ln_tile_dual(const float* C, int ldc, const float* w, const float* b,
                      _Float16* g1, int ldg, _Float16* o2, int ld2, float* red) {
  const int row = threadIdx.x >> 4, seg = threadIdx.x & 15;
  float mu; const float inv = ln_stats(C, ldc, red, row, seg, mu);
#pragma unroll
  for (int j = 0; j < 8; ++j) {
    const int c = seg * 8 + j;
    float v = tanhf((C[row * ldc + c] - mu) * inv * w[c] + b[c]);
    const _Float16 hv = (_Float16)v;
    g1[row * ldg + c] = hv;
    o2[row * ld2 + c] = hv;
  }
  __syncthreads();
}

struct TxW {
  const _Float16 *win, *wout, *wff1, *wff2;
  const float *bin, *bout, *bff1, *bff2, *n1w, *n1b, *n2w, *n2b;
};

DEV TxW glob_tx(const TxW& t) {
  TxW r;
  r.win = as_global(t.win); r.wout = as_global(t.wout);
  r.wff1 = as_global(t.wff1); r.wff2 = as_global(t.wff2);
  r.bin = as_global(t.bin); r.bout = as_global(t.bout);
  r.bff1 = as_global(t.bff1); r.bff2 = as_global(t.bff2);
  r.n1w = as_global(t.n1w); r.n1b = as_global(t.n1b);
  r.n2w = as_global(t.n2w); r.n2b = as_global(t.n2b);
  return r;
}

// Post-norm transformer encoder for S tokens x 16 nodes x 128, 1 head.
// X:[S][16][128] f16 (preserved). qkv:[S][16][384]; q-slot -> attn output,
// k-slot -> x1, v-slot -> final output pe. h: >= S*16*128 halves scratch.
template <int S>
DEV void encoder_block(_Float16* X, _Float16* qkv, _Float16* h,
                       float* C, float* red, float* aw, const TxW& t) {
  const int wv = threadIdx.x >> 5;
  for (int tI = wv; tI < S * 24; tI += 8) {
    const int s = tI / 24, n0 = (tI % 24) * 16;
    v8f acc = gemm16<128>(X + s * 2048, 128, t.win, n0);
    store_f16(qkv + s * 6144, 384, n0, acc, t.bin, false);
  }
  __syncthreads();
  // raw scores: one thread per (tt, s, i), 16B LDS reads.
  if (threadIdx.x < S * S * 16) {
    const int i = threadIdx.x & 15;
    const int s = (threadIdx.x >> 4) % S;
    const int tt = (threadIdx.x >> 4) / S;
    const _Float16* q = qkv + (s * 16 + i) * 384;
    const _Float16* k = qkv + (tt * 16 + i) * 384 + 128;
    float d = 0.f;
#pragma unroll
    for (int c8 = 0; c8 < 128; c8 += 8) {
      v8h qa = *(const v8h*)(q + c8);
      v8h kb = *(const v8h*)(k + c8);
#pragma unroll
      for (int j = 0; j < 8; ++j) d += (float)qa[j] * (float)kb[j];
    }
    aw[(i * S + s) * S + tt] = d * 0.088388347648318447f;  // 1/sqrt(128)
  }
  __syncthreads();
  // softmax over tt.
  if (threadIdx.x < S * 16) {
    const int s = threadIdx.x >> 4, i = threadIdx.x & 15;
    float sc[S]; float mx = -3.4e38f;
    for (int tt = 0; tt < S; ++tt) { sc[tt] = aw[(i * S + s) * S + tt]; mx = fmaxf(mx, sc[tt]); }
    float se = 0.f;
    for (int tt = 0; tt < S; ++tt) { sc[tt] = __expf(sc[tt] - mx); se += sc[tt]; }
    const float r = 1.f / se;
    for (int tt = 0; tt < S; ++tt) aw[(i * S + s) * S + tt] = sc[tt] * r;
  }
  __syncthreads();
  // o = attn @ v (into q-slot).
  for (int idx = threadIdx.x; idx < S * 2048; idx += 256) {
    const int c = idx & 127, i = (idx >> 7) & 15, s = idx >> 11;
    float a = 0.f;
    for (int tt = 0; tt < S; ++tt)
      a += aw[(i * S + s) * S + tt] * (float)qkv[(tt * 16 + i) * 384 + 256 + c];
    qkv[(s * 16 + i) * 384 + c] = (_Float16)a;
  }
  __syncthreads();
  // out-proj + residual + LN1 -> x1 (k-slot).
  for (int s = 0; s < S; ++s) {
    v8f acc = gemm16<128>(qkv + s * 6144, 384, t.wout, wv * 16);
    store_f32_res(C, 128, wv * 16, acc, t.bout, X + s * 2048, 128);
    __syncthreads();
    ln_tile_lds(C, 128, t.n1w, t.n1b, qkv + s * 6144 + 128, 384, false, red);
  }
  // ff1 + relu -> h.
  for (int tI = wv; tI < S * 8; tI += 8) {
    const int s = tI >> 3, n0 = (tI & 7) * 16;
    v8f acc = gemm16<128>(qkv + s * 6144 + 128, 384, t.wff1, n0);
    store_f16(h + s * 2048, 128, n0, acc, t.bff1, true);
  }
  __syncthreads();
  // ff2 + residual + LN2 -> pe (v-slot).
  for (int s = 0; s < S; ++s) {
    v8f acc = gemm16<128>(h + s * 2048, 128, t.wff2, wv * 16);
    store_f32_res(C, 128, wv * 16, acc, t.bff2, qkv + s * 6144 + 128, 384);
    __syncthreads();
    ln_tile_lds(C, 128, t.n2w, t.n2b, qkv + s * 6144 + 256, 384, false, red);
  }
}

// Shared embed step: src (f32, LDS or global per call site) -> tanh(LN(GEMM)).
template <int F>
DEV void embed_tile(const float* src, const _Float16* W, const float* B,
                    const float* lnw, const float* lnb, _Float16* dst,
                    _Float16* h, float* C, float* red) {
  for (int idx = threadIdx.x; idx < 16 * F; idx += 256)
    h[idx] = (_Float16)src[idx];
  __syncthreads();
  v8f acc = gemm16<F>(h, F, W, (threadIdx.x >> 5) * 16);
  store_f32(C, 128, (threadIdx.x >> 5) * 16, acc, B);
  __syncthreads();
  ln_tile_lds(C, 128, lnw, lnb, dst, 128, true, red);
}

// ------------------------------- kernels ----------------------------------

__global__ void cvt_f16_kernel(const float* s, _Float16* d, int n) {
  const int i = blockIdx.x * blockDim.x + threadIdx.x;
  if (i < n) as_global(d)[i] = (_Float16)as_global(s)[i];
}

__global__ void zero_f32_kernel(float* p, size_t n) {
  float* g = as_global(p);
  size_t i = (size_t)blockIdx.x * blockDim.x + threadIdx.x;
  const size_t stride = (size_t)gridDim.x * blockDim.x;
  for (; i < n; i += stride) g[i] = 0.f;
}

struct BranchArgs {
  const float *fc, *fe, *fm;
  const int* pair;
  const _Float16 *pwC, *pwE, *pwM;
  const float *pbC, *pbE, *pbM;
  const float *lnw, *lnb;
  TxW tx;
  const _Float16* pproj; const float* pprojb;
  const float *attw, *attb;
  _Float16* pathEmb; float* att; float* expsum;
};

__global__ __launch_bounds__(256) void branch_kernel(BranchArgs A) {
  extern __shared__ char smem[];
  _Float16* X   = (_Float16*)smem;       // [3][16][128] embeds (persist)
  _Float16* qkv = X + 3 * 2048;          // [3][16][384]
  _Float16* h   = qkv + 3 * 6144;        // [16][384] scratch
  float* C     = (float*)(h + 6144);     // [16][128]
  float* red   = C + 2048;               // 512
  float* aw    = red + 512;              // 144
  float* stage = aw + 144;               // 16*224 f32 feature staging (16B aligned)
  const int row0 = blockIdx.x * 16;
  const int wv = threadIdx.x >> 5;

  const float* fc = as_global(A.fc);
  const float* fe = as_global(A.fe);
  const float* fm = as_global(A.fm);
  const int* pair = as_global(A.pair);
  const _Float16* pw[3] = {as_global(A.pwC), as_global(A.pwE), as_global(A.pwM)};
  const float* pb[3] = {as_global(A.pbC), as_global(A.pbE), as_global(A.pbM)};
  const float* lnw = as_global(A.lnw);
  const float* lnb = as_global(A.lnb);
  const TxW tx = glob_tx(A.tx);
  const _Float16* pproj = as_global(A.pproj);
  const float* pprojb = as_global(A.pprojb);
  const float* attw = as_global(A.attw);
  const float* attb = as_global(A.attb);
  _Float16* pathEmb = as_global(A.pathEmb);
  float* att = as_global(A.att);
  float* expsum = as_global(A.expsum);

  // L2 prefetch for the next tile (global_prefetch_b8).
  if (blockIdx.x + 1 < gridDim.x && threadIdx.x < 16) {
    __builtin_prefetch(fc + (size_t)(row0 + 16 + threadIdx.x) * 128, 0, 1);
    __builtin_prefetch(fm + (size_t)(row0 + 16 + threadIdx.x) * 64, 0, 1);
  }

  // --- async-stage feature tiles (f32) into LDS via the async engine ---
  {
    const float* srcs[3] = {fc + (size_t)row0 * 128, fe + (size_t)row0 * 32,
                            fm + (size_t)row0 * 64};
    const int cnt16[3] = {512, 128, 256};   // 16-byte chunks per tile
    const int sOff[3]  = {0, 8192, 10240};  // byte offsets inside stage
    const unsigned stageBase = (unsigned)(unsigned long long)(const void*)stage;
    for (int ty = 0; ty < 3; ++ty) {
      for (int idx = threadIdx.x; idx < cnt16[ty]; idx += 256) {
        unsigned long long ga =
            (unsigned long long)(const void*)srcs[ty] + (unsigned long long)idx * 16;
        unsigned lo = stageBase + sOff[ty] + idx * 16;
        asm volatile("global_load_async_to_lds_b128 %0, %1, off"
                     :: "v"(lo), "v"(ga) : "memory");
      }
    }
    asm volatile("s_wait_asynccnt 0x0" ::: "memory");
  }
  __syncthreads();

  // --- embeds: tanh(LN(x @ W^T + b)) ---
  embed_tile<128>(stage,        pw[0], pb[0], lnw, lnb, X,            h, C, red);
  embed_tile<32> (stage + 2048, pw[1], pb[1], lnw, lnb, X + 2048,     h, C, red);
  embed_tile<64> (stage + 2560, pw[2], pb[2], lnw, lnb, X + 2 * 2048, h, C, red);

  encoder_block<3>(X, qkv, h, C, red, aw, tx);

  // --- path_proj over concat of the 3 encoded tokens ---
  for (int idx = threadIdx.x; idx < 16 * 384; idx += 256) {
    const int i = idx / 384, k = idx - i * 384;
    h[idx] = qkv[((k >> 7) * 16 + i) * 384 + 256 + (k & 127)];
  }
  __syncthreads();
  v8f acc = gemm16<384>(h, 384, pproj, wv * 16);
  store_f32(C, 128, wv * 16, acc, pprojb);
  __syncthreads();
  // LN + tanh -> global pathEmb (f16) + LDS copy (q-slot of token 0).
  ln_tile_dual(C, 128, lnw, lnb, pathEmb + (size_t)row0 * 128, 128, qkv, 384, red);

  // --- attention score: 16 lanes per row, shfl reduction ---
  {
    const int i = threadIdx.x >> 4, ln = threadIdx.x & 15;
    const _Float16* e0 = X + i * 128;
    const _Float16* pe = qkv + i * 384;
    const _Float16* e2 = X + 2 * 2048 + i * 128;
    float d = 0.f;
#pragma unroll
    for (int r = 0; r < 3; ++r) {
      const int c0 = (ln + 16 * r) * 8;
      const _Float16* srcp = (c0 < 128) ? (e0 + c0)
                           : (c0 < 256) ? (pe + (c0 - 128)) : (e2 + (c0 - 256));
      v8h v = *(const v8h*)srcp;
#pragma unroll
      for (int j = 0; j < 8; ++j) d += (float)v[j] * attw[c0 + j];
    }
#pragma unroll
    for (int m = 8; m >= 1; m >>= 1) d += __shfl_xor(d, m, 16);
    if (ln == 0) {
      const float a = tanhf(d + attb[0]);
      att[row0 + i] = a;
      atomicAdd(&expsum[pair[row0 + i]], __expf(a));  // tanh in (-1,1): no max needed
    }
  }
}

__global__ void aggregate_kernel(const _Float16* pathEmb_, const float* att_,
                                 const float* expsum_, const int* pair_, float* pp_) {
  const _Float16* pathEmb = as_global(pathEmb_);
  const float* att = as_global(att_);
  const float* expsum = as_global(expsum_);
  const int* pair = as_global(pair_);
  float* pp = as_global(pp_);
  const int node = blockIdx.x;
  const int c = threadIdx.x;
  const int p = pair[node];
  const float w = __expf(att[node]) / expsum[p];
  atomicAdd(&pp[(size_t)p * 128 + c], w * (float)pathEmb[(size_t)node * 128 + c]);
}

struct PairArgs {
  const float *pp0, *pp1;
  const float *company, *mobile;
  const _Float16 *pwC, *pwM;
  const float *pbC, *pbM;
  const float *lnw, *lnb;
  TxW tx;
  const float *predw, *predb;
  float* out;
};

__global__ __launch_bounds__(256) void pair_kernel(PairArgs P) {
  extern __shared__ char smem[];
  _Float16* X   = (_Float16*)smem;       // [2][16][128]
  _Float16* qkv = X + 2 * 2048;          // [2][16][384]
  _Float16* h   = qkv + 2 * 6144;        // [16][384] scratch
  float* C   = (float*)(h + 6144);
  float* red = C + 2048;
  float* aw  = red + 512;
  const int row0 = blockIdx.x * 16;
  const int wv = threadIdx.x >> 5;

  const float* pp0 = as_global(P.pp0);
  const float* pp1 = as_global(P.pp1);
  const float* company = as_global(P.company);
  const float* mobile = as_global(P.mobile);
  const _Float16* pwC = as_global(P.pwC);
  const _Float16* pwM = as_global(P.pwM);
  const float* pbC = as_global(P.pbC);
  const float* pbM = as_global(P.pbM);
  const float* lnw = as_global(P.lnw);
  const float* lnb = as_global(P.lnb);
  const TxW tx = glob_tx(P.tx);
  const float* predw = as_global(P.predw);
  const float* predb = as_global(P.predb);
  float* out = as_global(P.out);

  for (int idx = threadIdx.x; idx < 2 * 2048; idx += 256) {
    const int s = idx >> 11, rem = idx & 2047;
    const float* src = (s == 0 ? pp0 : pp1) + (size_t)row0 * 128 + rem;
    X[idx] = (_Float16)(*src);
  }
  __syncthreads();

  encoder_block<2>(X, qkv, h, C, red, aw, tx);

  // y = tanh(LN(pe)) per token -> X (embeds no longer needed).
  for (int s = 0; s < 2; ++s) {
    for (int idx = threadIdx.x; idx < 2048; idx += 256)
      C[idx] = (float)qkv[(s * 16 + (idx >> 7)) * 384 + 256 + (idx & 127)];
    __syncthreads();
    ln_tile_lds(C, 128, lnw, lnb, X + s * 2048, 128, true, red);
  }

  // start/end embeds (h region: startE | endE | featA).
  _Float16* startE = h;
  _Float16* endE   = h + 2048;
  embed_tile<128>(company + (size_t)row0 * 128, pwC, pbC, lnw, lnb, startE,
                  h + 4096, C, red);
  embed_tile<64> (mobile + (size_t)row0 * 64,  pwM, pbM, lnw, lnb, endE,
                  h + 4096, C, red);

  // pred: sigmoid([start|end|y0|y1] @ w^T + b), 16 lanes per row + shfl.
  {
    const int i = threadIdx.x >> 4, ln = threadIdx.x & 15;
    float z = 0.f;
#pragma unroll
    for (int r = 0; r < 4; ++r) {
      const int c0 = (ln + 16 * r) * 8;
      const _Float16* srcp =
          (c0 < 128) ? (startE + i * 128 + c0)
        : (c0 < 256) ? (endE + i * 128 + (c0 - 128))
        : (c0 < 384) ? (X + i * 128 + (c0 - 256))
                     : (X + 2048 + i * 128 + (c0 - 384));
      v8h v = *(const v8h*)srcp;
#pragma unroll
      for (int j = 0; j < 8; ++j) z += (float)v[j] * predw[c0 + j];
    }
#pragma unroll
    for (int m = 8; m >= 1; m >>= 1) z += __shfl_xor(z, m, 16);
    if (ln == 0) out[row0 + i] = 1.f / (1.f + __expf(-(z + predb[0])));
  }
}

// ------------------------------ host side ----------------------------------

extern "C" void kernel_launch(void* const* d_in, const int* in_sizes, int n_in,
                              void* d_out, int out_size, void* d_ws, size_t ws_size,
                              hipStream_t stream) {
  const float* company = (const float*)d_in[0];
  const float* mobile  = (const float*)d_in[1];
  const float* pfeat[2][3] = {
    {(const float*)d_in[2], (const float*)d_in[3], (const float*)d_in[4]},
    {(const float*)d_in[5], (const float*)d_in[6], (const float*)d_in[7]}};
  const int* ppair[2] = {(const int*)d_in[8], (const int*)d_in[9]};
  const int N = in_sizes[8];
  const int PAIRS = in_sizes[0] / 128;

  const float* projW[3] = {(const float*)d_in[10], (const float*)d_in[12], (const float*)d_in[14]};
  const float* projB[3] = {(const float*)d_in[11], (const float*)d_in[13], (const float*)d_in[15]};
  const float* lnw = (const float*)d_in[16];
  const float* lnb = (const float*)d_in[17];
  const int txBase[3] = {18, 30, 50};  // path_tx0, path_tx1, pair_tx
  auto txF = [&](int base, int j) { return (const float*)d_in[base + j]; };
  const float* pprojW[2] = {(const float*)d_in[42], (const float*)d_in[44]};
  const float* pprojB[2] = {(const float*)d_in[43], (const float*)d_in[45]};
  const float* attW[2]   = {(const float*)d_in[46], (const float*)d_in[48]};
  const float* attB[2]   = {(const float*)d_in[47], (const float*)d_in[49]};
  const float* predW = (const float*)d_in[62];
  const float* predB = (const float*)d_in[63];

  // ---- workspace carve (~156 MB) ----
  char* w = (char*)d_ws;
  auto carve = [&](size_t bytes) -> void* {
    void* p = (void*)w; w += (bytes + 255) & ~(size_t)255; return p;
  };
  const int finArr[3] = {128, 32, 64};
  _Float16* projW16[3];
  for (int t = 0; t < 3; ++t) projW16[t] = (_Float16*)carve((size_t)128 * finArr[t] * 2);
  const size_t txN[4] = {384 * 128, 128 * 128, 128 * 128, 128 * 128};
  _Float16* txW16[3][4];
  for (int t = 0; t < 3; ++t)
    for (int j = 0; j < 4; ++j) txW16[t][j] = (_Float16*)carve(txN[j] * 2);
  _Float16* pprojW16[2];
  for (int t = 0; t < 2; ++t) pprojW16[t] = (_Float16*)carve((size_t)128 * 384 * 2);
  _Float16* pathEmb[2];
  for (int t = 0; t < 2; ++t) pathEmb[t] = (_Float16*)carve((size_t)N * 128 * 2);
  float* attBuf[2];
  for (int t = 0; t < 2; ++t) attBuf[t] = (float*)carve((size_t)N * 4);
  float* zeroBase = (float*)(void*)w;
  float* expsum[2];
  for (int t = 0; t < 2; ++t) expsum[t] = (float*)carve((size_t)PAIRS * 4);
  float* pp[2];
  for (int t = 0; t < 2; ++t) pp[t] = (float*)carve((size_t)PAIRS * 128 * 4);
  const size_t zeroCount = (size_t)((float*)(void*)w - zeroBase);

  auto cvt = [&](const float* s, _Float16* d, size_t n) {
    cvt_f16_kernel<<<dim3((unsigned)((n + 255) / 256)), dim3(256), 0, stream>>>(s, d, (int)n);
  };
  for (int t = 0; t < 3; ++t) cvt(projW[t], projW16[t], (size_t)128 * finArr[t]);
  const int txWidx[4] = {0, 2, 4, 6};  // in_proj.w, out.w, ff1.w, ff2.w
  for (int t = 0; t < 3; ++t)
    for (int j = 0; j < 4; ++j) cvt(txF(txBase[t], txWidx[j]), txW16[t][j], txN[j]);
  for (int t = 0; t < 2; ++t) cvt(pprojW[t], pprojW16[t], (size_t)128 * 384);

  zero_f32_kernel<<<dim3(2048), dim3(256), 0, stream>>>(zeroBase, zeroCount);

  auto mkTx = [&](int t) {
    TxW x; const int b = txBase[t];
    x.win = txW16[t][0]; x.wout = txW16[t][1]; x.wff1 = txW16[t][2]; x.wff2 = txW16[t][3];
    x.bin = txF(b, 1); x.bout = txF(b, 3); x.bff1 = txF(b, 5); x.bff2 = txF(b, 7);
    x.n1w = txF(b, 8); x.n1b = txF(b, 9); x.n2w = txF(b, 10); x.n2b = txF(b, 11);
    return x;
  };

  // LDS: halves 6144+18432+6144 -> 61440B; floats C/red/aw 10816B; stage 14336B.
  const size_t shB = (size_t)(3 * 2048 + 3 * 6144 + 6144) * 2 +
                     (2048 + 512 + 144) * 4 + (size_t)16 * 224 * 4;
  for (int br = 0; br < 2; ++br) {
    BranchArgs A;
    A.fc = pfeat[br][0]; A.fe = pfeat[br][1]; A.fm = pfeat[br][2];
    A.pair = ppair[br];
    A.pwC = projW16[0]; A.pwE = projW16[1]; A.pwM = projW16[2];
    A.pbC = projB[0]; A.pbE = projB[1]; A.pbM = projB[2];
    A.lnw = lnw; A.lnb = lnb;
    A.tx = mkTx(br);
    A.pproj = pprojW16[br]; A.pprojb = pprojB[br];
    A.attw = attW[br]; A.attb = attB[br];
    A.pathEmb = pathEmb[br]; A.att = attBuf[br]; A.expsum = expsum[br];
    branch_kernel<<<dim3(N / 16), dim3(256), shB, stream>>>(A);
  }

  for (int br = 0; br < 2; ++br)
    aggregate_kernel<<<dim3(N), dim3(128), 0, stream>>>(
        pathEmb[br], attBuf[br], expsum[br], ppair[br], pp[br]);

  PairArgs P;
  P.pp0 = pp[0]; P.pp1 = pp[1];
  P.company = company; P.mobile = mobile;
  P.pwC = projW16[0]; P.pwM = projW16[2];
  P.pbC = projB[0]; P.pbM = projB[2];
  P.lnw = lnw; P.lnb = lnb;
  P.tx = mkTx(2);
  P.predw = predW; P.predb = predB;
  P.out = (float*)d_out;
  const size_t shP = (size_t)(2 * 2048 + 2 * 6144 + 6144) * 2 + (2048 + 512 + 144) * 4;
  pair_kernel<<<dim3(PAIRS / 16), dim3(256), shP, stream>>>(P);
}
